// SingleHeadLongformerModel_18794776887866
// MI455X (gfx1250) — matrix operand order repaired and checked by reading, not compiled, not run on hardware
//
#include <hip/hip_runtime.h>
#include <hip/hip_bf16.h>
#include <math.h>

// ---------------------------------------------------------------------------
// Longformer-style encoder forward for MI455X (gfx1250, wave32, WMMA).
// All dense math goes through v_wmma_f32_16x16x32_bf16 (bf16 in, f32 acc).
// ---------------------------------------------------------------------------

#define S_LEN   4096
#define DHID    768
#define NHEAD   12
#define DHEAD   64
#define DFF_    3072
#define NLAYER  12
#define WIN     256
#define NCHUNK  16          // S_LEN / WIN
#define ATT_SCALE 0.125f    // 1/sqrt(64)

typedef __attribute__((ext_vector_type(16))) __bf16 v16bf;
typedef __attribute__((ext_vector_type(8)))  float  v8f;

union BF16x16 { v16bf v; uint4 q[2]; __bf16 e[16]; };
union BF16x4  { __bf16 e[4]; uint2 u; };

// ======================== generic WMMA GEMM ================================
// C[M,N] = epilogue(A[M,K] @ B[K,N] + bias), A/B fp32 in memory, bf16 in LDS.
// Block tile 128(M) x 128(N), K-step 32. 256 threads = 8 waves; each wave
// owns a 16(M) x 128(N) strip -> 8 accumulators of 16x16.
// Requires M % 128 == 0, N % 128 == 0, K % 32 == 0 (true for all uses here).
// epi: 0=bias  1=(..)*scale  2=GELU(exact)  3=+residual
#define BM 128
#define BN 128
#define BK 32

__global__ __launch_bounds__(256) void gemm_bf16_wmma(
    const float* __restrict__ A, const float* __restrict__ B,
    const float* __restrict__ bias, const float* __restrict__ res,
    float* __restrict__ Cout, int M, int N, int K, int epi, float scale)
{
  __shared__ __align__(32) __bf16 As[BM][BK];     // row-major
  __shared__ __align__(32) __bf16 Bs[BN][BK];     // transposed: Bs[n][k]

  const int t    = threadIdx.x;
  const int lane = t & 31;
  const int w    = t >> 5;         // wave 0..7 : rows w*16 .. w*16+15
  const int m0   = blockIdx.y * BM;
  const int n0   = blockIdx.x * BN;

  v8f acc[8];
  #pragma unroll
  for (int i = 0; i < 8; ++i)
    #pragma unroll
    for (int r = 0; r < 8; ++r) acc[i][r] = 0.0f;

  // staging assignments (vectorized: 16 contiguous fp32 per thread per tile)
  const int arow = t >> 1;            // 0..127
  const int aks  = (t & 1) * 16;      // 0 or 16
  const int bkk  = t >> 3;            // 0..31
  const int bns  = (t & 7) * 16;      // 0..112

  for (int k0 = 0; k0 < K; k0 += BK) {
    // ---- stage A tile (128x32): float4 loads, packed bf16 b128 stores ----
    {
      const float4* ga = (const float4*)(A + (size_t)(m0 + arow) * K + k0 + aks);
      float4 f0 = ga[0], f1 = ga[1], f2 = ga[2], f3 = ga[3];
      BF16x16 pk;
      pk.e[0]=(__bf16)f0.x;  pk.e[1]=(__bf16)f0.y;  pk.e[2]=(__bf16)f0.z;  pk.e[3]=(__bf16)f0.w;
      pk.e[4]=(__bf16)f1.x;  pk.e[5]=(__bf16)f1.y;  pk.e[6]=(__bf16)f1.z;  pk.e[7]=(__bf16)f1.w;
      pk.e[8]=(__bf16)f2.x;  pk.e[9]=(__bf16)f2.y;  pk.e[10]=(__bf16)f2.z; pk.e[11]=(__bf16)f2.w;
      pk.e[12]=(__bf16)f3.x; pk.e[13]=(__bf16)f3.y; pk.e[14]=(__bf16)f3.z; pk.e[15]=(__bf16)f3.w;
      uint4* da = (uint4*)&As[arow][aks];
      da[0] = pk.q[0];
      da[1] = pk.q[1];
    }
    // ---- stage B tile (32x128) transposed into Bs[n][k] ----
    {
      const float4* gb = (const float4*)(B + (size_t)(k0 + bkk) * N + n0 + bns);
      float4 f0 = gb[0], f1 = gb[1], f2 = gb[2], f3 = gb[3];
      __bf16 e[16];
      e[0]=(__bf16)f0.x;  e[1]=(__bf16)f0.y;  e[2]=(__bf16)f0.z;  e[3]=(__bf16)f0.w;
      e[4]=(__bf16)f1.x;  e[5]=(__bf16)f1.y;  e[6]=(__bf16)f1.z;  e[7]=(__bf16)f1.w;
      e[8]=(__bf16)f2.x;  e[9]=(__bf16)f2.y;  e[10]=(__bf16)f2.z; e[11]=(__bf16)f2.w;
      e[12]=(__bf16)f3.x; e[13]=(__bf16)f3.y; e[14]=(__bf16)f3.z; e[15]=(__bf16)f3.w;
      #pragma unroll
      for (int j = 0; j < 16; ++j) Bs[bns + j][bkk] = e[j];
    }
    // prefetch next tiles into cache (global_prefetch_b8)
    if (k0 + BK < K) {
      __builtin_prefetch(&A[(size_t)(m0 + (t >> 1)) * K + (k0 + BK)], 0, 0);
      __builtin_prefetch(&B[(size_t)(k0 + BK + (t >> 3)) * N + n0 + ((t & 7) << 4)], 0, 0);
    }
    __syncthreads();

    // ---- compute: one A fragment per wave, 8 B fragments ----
    const int mrow = w * 16 + (lane & 15);
    const int kb   = (lane >> 4) * 8;   // unused; A and B frags both use kb2
    const int kb2  = (lane >> 4) * 16;  // A and B frag K base (contiguous k = 16*(lane/16) + s)
    BF16x16 a;
    a.q[0] = *(const uint4*)&As[mrow][kb2];
    a.q[1] = *(const uint4*)&As[mrow][kb2 + 8];
    #pragma unroll
    for (int i = 0; i < 8; ++i) {
      int col = i * 16 + (lane & 15);
      BF16x16 b;
      b.v = *(const v16bf*)&Bs[col][kb2];
      acc[i] = __builtin_amdgcn_wmma_f32_16x16x32_bf16(
          false, a.v, false, b.v, (short)0, acc[i], false, false);
    }
    __syncthreads();
  }

  // ---- epilogue (C layout: lane%16 = col, vgpr r -> row r + 8*(lane/16)) ----
  #pragma unroll
  for (int i = 0; i < 8; ++i) {
    int col = n0 + i * 16 + (lane & 15);
    float bv = bias ? bias[col] : 0.0f;
    #pragma unroll
    for (int r = 0; r < 8; ++r) {
      int row = m0 + w * 16 + r + ((lane >> 4) << 3);
      size_t idx = (size_t)row * N + col;
      float vv = acc[i][r] + bv;
      if (epi == 1)      vv *= scale;
      else if (epi == 2) vv = 0.5f * vv * (1.0f + erff(vv * 0.70710678118654752f));
      else if (epi == 3) vv += res[idx];
      Cout[idx] = vv;
    }
  }
}

// ===================== fused sliding-window attention ======================
// One block per (head, chunk, 16-query tile). 128 threads = 4 waves.
// Phase A: scores = Q(16x64) @ K^T(64x768) via WMMA into LDS fp32.
// Phase B: band/mask softmax over 768 window keys + 1 global key
//          (8 threads per row), probabilities written as bf16 tile Pb.
// Phase C: out = P(16x768) @ V(768x64) via WMMA + p_global * v[0].
__global__ __launch_bounds__(128) void attn_window(
    const float* __restrict__ q, const float* __restrict__ k,
    const float* __restrict__ v, const int* __restrict__ amask,
    float* __restrict__ out)
{
  __shared__ __align__(32) __bf16 Qs[16][64];       // 2 KB
  __shared__ __align__(32) __bf16 KVs[64 * 256];    // 32 KB (K tile, then V^T tile)
  __shared__ __align__(32) __bf16 Pb[16][3 * WIN];  // 24 KB probabilities (bf16)
  __shared__ float Sc[16][3 * WIN];                 // 48 KB scores scratch
  __shared__ float Pg[16];
  __shared__ float mxs[16], invs[16], gsv[16];
  __shared__ float rred[16][8];

  const int t    = threadIdx.x;
  const int lane = t & 31;
  const int w    = t >> 5;               // wave 0..3
  const int bid  = blockIdx.x;
  const int m16  = bid & 15;
  const int c    = (bid >> 4) & 15;
  const int h    = bid >> 8;
  const int qr0  = c * WIN + m16 * 16;
  const int hoff = h * DHEAD;

  // ---- stage Q tile (16x64) vectorized ----
  #pragma unroll
  for (int i = 0; i < 2; ++i) {
    int qd = t + 128 * i;               // 0..255 quads
    int r  = qd >> 4;
    int d4 = (qd & 15) << 2;
    const float4 f = *(const float4*)(q + (size_t)(qr0 + r) * DHID + hoff + d4);
    BF16x4 pk;
    pk.e[0]=(__bf16)f.x; pk.e[1]=(__bf16)f.y; pk.e[2]=(__bf16)f.z; pk.e[3]=(__bf16)f.w;
    *(uint2*)&Qs[r][d4] = pk.u;
  }

  const int kb  = (lane >> 4) * 8;
  const int kb2 = (lane >> 4) * 16;

  // ---------------- Phase A: scores ----------------
  for (int kc = 0; kc < 3; ++kc) {
    __syncthreads();
    for (int i = 0; i < 32; ++i) {
      int qd = t + 128 * i;             // 0..4095 quads
      int j  = qd >> 4;                 // key within chunk
      int d4 = (qd & 15) << 2;
      int kpos = (c - 1) * WIN + kc * 256 + j;
      BF16x4 pk;
      if (kpos >= 0 && kpos < S_LEN) {
        const float4 f = *(const float4*)(k + (size_t)kpos * DHID + hoff + d4);
        pk.e[0]=(__bf16)f.x; pk.e[1]=(__bf16)f.y; pk.e[2]=(__bf16)f.z; pk.e[3]=(__bf16)f.w;
      } else {
        pk.u.x = 0u; pk.u.y = 0u;
      }
      *(uint2*)&KVs[j * 64 + d4] = pk.u;  // Kt[key][d]
    }
    __syncthreads();
    #pragma unroll
    for (int nt = 0; nt < 4; ++nt) {
      int jBase = (w * 4 + nt) * 16;
      v8f sacc = {0.f,0.f,0.f,0.f,0.f,0.f,0.f,0.f};
      #pragma unroll
      for (int kk0 = 0; kk0 < 64; kk0 += 32) {
        int row = lane & 15;
        BF16x16 a, b;
        a.q[0] = *(const uint4*)&Qs[row][kk0 + kb2];
        a.q[1] = *(const uint4*)&Qs[row][kk0 + kb2 + 8];
        int col = jBase + (lane & 15);
        b.v = *(const v16bf*)&KVs[col * 64 + kk0 + kb2];
        sacc = __builtin_amdgcn_wmma_f32_16x16x32_bf16(
            false, a.v, false, b.v, (short)0, sacc, false, false);
      }
      int colS = kc * 256 + jBase + (lane & 15);
      #pragma unroll
      for (int r = 0; r < 8; ++r)
        Sc[r + ((lane >> 4) << 3)][colS] = sacc[r];
    }
  }
  __syncthreads();

  // ---------------- Phase B: masked softmax (8 threads per row) ----------------
  const int row8 = t >> 3;     // 0..15
  const int sub  = t & 7;      // 0..7
  const bool gm  = amask[0] != 0;
  if (sub == 0) {
    float gs = 0.0f;
    #pragma unroll 8
    for (int d = 0; d < DHEAD; ++d) gs += (float)Qs[row8][d] * k[hoff + d];
    gsv[row8] = gs;
  }
  __syncthreads();
  {
    const int qpos = qr0 + row8;
    float lmax = -INFINITY;
    for (int j = sub; j < 3 * WIN; j += 8) {
      int kpos = (c - 1) * WIN + j;
      int dd = kpos - qpos; if (dd < 0) dd = -dd;
      bool inr  = (kpos > 0) && (kpos < S_LEN) && (dd <= WIN);
      bool band = inr && (amask[inr ? kpos : 0] != 0);
      float sv = band ? Sc[row8][j] : -INFINITY;
      Sc[row8][j] = sv;
      lmax = fmaxf(lmax, sv);
    }
    rred[row8][sub] = lmax;
  }
  __syncthreads();
  if (sub == 0) {
    float m = gm ? gsv[row8] : -INFINITY;
    #pragma unroll
    for (int i = 0; i < 8; ++i) m = fmaxf(m, rred[row8][i]);
    mxs[row8] = m;
  }
  __syncthreads();
  {
    const float mx = mxs[row8];
    float lsum = 0.0f;
    for (int j = sub; j < 3 * WIN; j += 8) {
      float ee = __expf(Sc[row8][j] - mx);  // exp(-inf)=0 for masked
      Sc[row8][j] = ee;
      lsum += ee;
    }
    rred[row8][sub] = lsum;
  }
  __syncthreads();
  if (sub == 0) {
    float s = 0.0f;
    #pragma unroll
    for (int i = 0; i < 8; ++i) s += rred[row8][i];
    float eg = gm ? __expf(gsv[row8] - mxs[row8]) : 0.0f;
    s += eg;
    float inv = 1.0f / s;
    invs[row8] = inv;
    Pg[row8] = eg * inv;
  }
  __syncthreads();
  {
    const float inv = invs[row8];
    for (int j = sub; j < 3 * WIN; j += 8)
      Pb[row8][j] = (__bf16)(Sc[row8][j] * inv);
  }

  // ---------------- Phase C: out = P @ V ----------------
  v8f oacc = {0.f,0.f,0.f,0.f,0.f,0.f,0.f,0.f};
  const int dBase = w * 16;
  const int rowA  = lane & 15;
  for (int kc = 0; kc < 3; ++kc) {
    __syncthreads();
    for (int i = 0; i < 32; ++i) {
      int qd = t + 128 * i;
      int j  = qd >> 4;
      int d4 = (qd & 15) << 2;
      int kpos = (c - 1) * WIN + kc * 256 + j;
      float4 f;
      if (kpos >= 0 && kpos < S_LEN) {
        f = *(const float4*)(v + (size_t)kpos * DHID + hoff + d4);
      } else {
        f.x = f.y = f.z = f.w = 0.0f;
      }
      KVs[(d4 + 0) * 256 + j] = (__bf16)f.x;   // Vt[d][key]
      KVs[(d4 + 1) * 256 + j] = (__bf16)f.y;
      KVs[(d4 + 2) * 256 + j] = (__bf16)f.z;
      KVs[(d4 + 3) * 256 + j] = (__bf16)f.w;
    }
    __syncthreads();
    #pragma unroll
    for (int kk = 0; kk < 256; kk += 32) {
      int base = kc * 256 + kk;
      BF16x16 a, b;
      a.q[0] = *(const uint4*)&Pb[rowA][base + kb2];
      a.q[1] = *(const uint4*)&Pb[rowA][base + kb2 + 8];
      int col = dBase + (lane & 15);
      b.v = *(const v16bf*)&KVs[col * 256 + kk + kb2];
      oacc = __builtin_amdgcn_wmma_f32_16x16x32_bf16(
          false, a.v, false, b.v, (short)0, oacc, false, false);
    }
  }
  __syncthreads();
  {
    int dcol = dBase + (lane & 15);
    float v0 = v[hoff + dcol];            // v at sequence position 0
    #pragma unroll
    for (int r = 0; r < 8; ++r) {
      int row = r + ((lane >> 4) << 3);
      out[(size_t)(qr0 + row) * DHID + hoff + dcol] = oacc[r] + Pg[row] * v0;
    }
  }
}

// ===================== global-token attention (row 0) ======================
__global__ __launch_bounds__(256) void attn_global(
    const float* __restrict__ x, const float* __restrict__ Wqg,
    const float* __restrict__ bqg, const float* __restrict__ kg,
    const float* __restrict__ vg, const int* __restrict__ amask,
    float* __restrict__ out)
{
  __shared__ float qg[DHEAD];
  __shared__ float sc[S_LEN];       // 16 KB
  __shared__ float red[256];
  __shared__ float gpart[4][DHEAD];

  const int t = threadIdx.x, h = blockIdx.x, hoff = h * DHEAD;

  if (t < DHEAD) {
    float a = bqg[hoff + t];
    for (int i = 0; i < DHID; ++i) a += x[i] * Wqg[(size_t)i * DHID + hoff + t];
    qg[t] = a * ATT_SCALE;
  }
  __syncthreads();

  float lmax = -INFINITY;
  for (int s = t; s < S_LEN; s += 256) {
    float a = 0.0f;
    #pragma unroll 8
    for (int d = 0; d < DHEAD; ++d) a += qg[d] * kg[(size_t)s * DHID + hoff + d];
    a = amask[s] ? a : -INFINITY;
    sc[s] = a;
    lmax = fmaxf(lmax, a);
  }
  red[t] = lmax; __syncthreads();
  for (int o = 128; o > 0; o >>= 1) { if (t < o) red[t] = fmaxf(red[t], red[t + o]); __syncthreads(); }
  float mx = red[0]; __syncthreads();

  float lsum = 0.0f;
  for (int s = t; s < S_LEN; s += 256) {
    float e = __expf(sc[s] - mx);
    sc[s] = e;
    lsum += e;
  }
  red[t] = lsum; __syncthreads();
  for (int o = 128; o > 0; o >>= 1) { if (t < o) red[t] += red[t + o]; __syncthreads(); }
  float inv = 1.0f / red[0]; __syncthreads();

  int g = t >> 6, d = t & 63;
  float part = 0.0f;
  for (int s = g; s < S_LEN; s += 4) part += sc[s] * vg[(size_t)s * DHID + hoff + d];
  gpart[g][d] = part;
  __syncthreads();
  if (t < DHEAD)
    out[hoff + t] = (gpart[0][t] + gpart[1][t] + gpart[2][t] + gpart[3][t]) * inv;
}

// ============================ LayerNorm ====================================
__global__ __launch_bounds__(256) void layernorm_k(
    const float* __restrict__ in, const float* __restrict__ sw,
    const float* __restrict__ bw, float* __restrict__ out)
{
  __shared__ float red[256];
  const int row = blockIdx.x, t = threadIdx.x;
  const float* p = in + (size_t)row * DHID;
  float s0 = 0.f, s1 = 0.f;
  for (int d = t; d < DHID; d += 256) { float vv = p[d]; s0 += vv; s1 += vv * vv; }
  red[t] = s0; __syncthreads();
  for (int o = 128; o > 0; o >>= 1) { if (t < o) red[t] += red[t + o]; __syncthreads(); }
  float mean = red[0] * (1.0f / DHID); __syncthreads();
  red[t] = s1; __syncthreads();
  for (int o = 128; o > 0; o >>= 1) { if (t < o) red[t] += red[t + o]; __syncthreads(); }
  float var = red[0] * (1.0f / DHID) - mean * mean;
  float inv = rsqrtf(var + 1e-5f);
  float* o = out + (size_t)row * DHID;
  for (int d = t; d < DHID; d += 256) o[d] = (p[d] - mean) * inv * sw[d] + bw[d];
}

// ============================ embedding ====================================
__global__ __launch_bounds__(256) void embed_k(
    const int* __restrict__ ids, const float* __restrict__ we,
    const float* __restrict__ pe, const float* __restrict__ tte,
    float* __restrict__ out)
{
  const int s = blockIdx.x, t = threadIdx.x;
  const int id = ids[s];
  for (int d = t; d < DHID; d += 256)
    out[(size_t)s * DHID + d] = we[(size_t)id * DHID + d] + pe[(size_t)(s + 2) * DHID + d] + tte[d];
}

// ============================== head =======================================
__global__ __launch_bounds__(512) void head_k(
    const float* __restrict__ x, const float* __restrict__ W1,
    const float* __restrict__ b1, const float* __restrict__ W2,
    const float* __restrict__ b2, float* __restrict__ out)
{
  __shared__ float red[512];
  const int t = threadIdx.x;
  float a = b1[t];
  for (int i = 0; i < DHID; ++i) a += x[i] * W1[(size_t)i * 512 + t];
  a = fmaxf(a, 0.0f);
  red[t] = a * W2[t];
  __syncthreads();
  for (int o = 256; o > 0; o >>= 1) { if (t < o) red[t] += red[t + o]; __syncthreads(); }
  if (t == 0) out[0] = red[0] + b2[0];
}

// ============================ launcher =====================================
extern "C" void kernel_launch(void* const* d_in, const int* in_sizes, int n_in,
                              void* d_out, int out_size, void* d_ws, size_t ws_size,
                              hipStream_t stream)
{
  (void)in_sizes; (void)n_in; (void)out_size; (void)ws_size;

  const int*   ids   = (const int*)d_in[0];
  const int*   amask = (const int*)d_in[1];
  const float* we    = (const float*)d_in[2];
  const float* pe    = (const float*)d_in[3];
  const float* tte   = (const float*)d_in[4];
  const float* lnes  = (const float*)d_in[5];
  const float* lneb  = (const float*)d_in[6];
  const float* Wq  = (const float*)d_in[7];   const float* bq  = (const float*)d_in[8];
  const float* Wk  = (const float*)d_in[9];   const float* bk  = (const float*)d_in[10];
  const float* Wv  = (const float*)d_in[11];  const float* bv  = (const float*)d_in[12];
  const float* Wo  = (const float*)d_in[13];  const float* bo  = (const float*)d_in[14];
  const float* Wqg = (const float*)d_in[15];  const float* bqg = (const float*)d_in[16];
  const float* Wkg = (const float*)d_in[17];  const float* bkg = (const float*)d_in[18];
  const float* Wvg = (const float*)d_in[19];  const float* bvg = (const float*)d_in[20];
  const float* ln1s= (const float*)d_in[21];  const float* ln1b= (const float*)d_in[22];
  const float* Wf1 = (const float*)d_in[23];  const float* bf1 = (const float*)d_in[24];
  const float* Wf2 = (const float*)d_in[25];  const float* bf2 = (const float*)d_in[26];
  const float* ln2s= (const float*)d_in[27];  const float* ln2b= (const float*)d_in[28];
  const float* Wh1 = (const float*)d_in[29];  const float* bh1 = (const float*)d_in[30];
  const float* Wh2 = (const float*)d_in[31];  const float* bh2 = (const float*)d_in[32];

  float* ws = (float*)d_ws;
  const size_t SD = (size_t)S_LEN * DHID;
  float* X  = ws + 0 * SD;   // hidden state
  float* Y  = ws + 1 * SD;   // pre-LN temp
  float* Hh = ws + 2 * SD;   // post-LN1 hidden
  float* Q  = ws + 3 * SD;
  float* Kb = ws + 4 * SD;
  float* Vb = ws + 5 * SD;
  float* KG = ws + 6 * SD;
  float* VG = ws + 7 * SD;
  float* AT = ws + 8 * SD;   // attention output
  float* FF = ws + 9 * SD;   // S x DFF

  dim3 blk(256);
  dim3 gN768(DHID / BN, S_LEN / BM);   // 6 x 32
  dim3 gN3072(DFF_ / BN, S_LEN / BM);  // 24 x 32

  embed_k<<<S_LEN, 256, 0, stream>>>(ids, we, pe, tte, Y);
  layernorm_k<<<S_LEN, 256, 0, stream>>>(Y, lnes, lneb, X);

  for (int l = 0; l < NLAYER; ++l) {
    const size_t wOff  = (size_t)l * DHID * DHID;
    const size_t bOff  = (size_t)l * DHID;
    const size_t w1Off = (size_t)l * DHID * DFF_;
    const size_t b1Off = (size_t)l * DFF_;

    gemm_bf16_wmma<<<gN768, blk, 0, stream>>>(X, Wq + wOff, bq + bOff, nullptr, Q,
                                              S_LEN, DHID, DHID, 1, ATT_SCALE);
    gemm_bf16_wmma<<<gN768, blk, 0, stream>>>(X, Wk + wOff, bk + bOff, nullptr, Kb,
                                              S_LEN, DHID, DHID, 0, 0.f);
    gemm_bf16_wmma<<<gN768, blk, 0, stream>>>(X, Wv + wOff, bv + bOff, nullptr, Vb,
                                              S_LEN, DHID, DHID, 0, 0.f);
    gemm_bf16_wmma<<<gN768, blk, 0, stream>>>(X, Wkg + wOff, bkg + bOff, nullptr, KG,
                                              S_LEN, DHID, DHID, 0, 0.f);
    gemm_bf16_wmma<<<gN768, blk, 0, stream>>>(X, Wvg + wOff, bvg + bOff, nullptr, VG,
                                              S_LEN, DHID, DHID, 0, 0.f);

    attn_window<<<NHEAD * NCHUNK * (WIN / 16), 128, 0, stream>>>(Q, Kb, Vb, amask, AT);
    attn_global<<<NHEAD, 256, 0, stream>>>(X, Wqg + wOff, bqg + bOff, KG, VG, amask, AT);

    gemm_bf16_wmma<<<gN768, blk, 0, stream>>>(AT, Wo + wOff, bo + bOff, X, Y,
                                              S_LEN, DHID, DHID, 3, 0.f);
    layernorm_k<<<S_LEN, 256, 0, stream>>>(Y, ln1s + bOff, ln1b + bOff, Hh);

    gemm_bf16_wmma<<<gN3072, blk, 0, stream>>>(Hh, Wf1 + w1Off, bf1 + b1Off, nullptr, FF,
                                               S_LEN, DFF_, DHID, 2, 0.f);
    gemm_bf16_wmma<<<gN768, blk, 0, stream>>>(FF, Wf2 + w1Off, bf2 + bOff, Hh, Y,
                                              S_LEN, DHID, DFF_, 3, 0.f);
    layernorm_k<<<S_LEN, 256, 0, stream>>>(Y, ln2s + bOff, ln2b + bOff, X);
  }

  head_k<<<1, 512, 0, stream>>>(X, Wh1, bh1, Wh2, bh2, (float*)d_out);
}
